// EMPSN_md17_2774548873304
// MI455X (gfx1250) — matrix operand order repaired and checked by reading, not compiled, not run on hardware
//
#include <hip/hip_runtime.h>
#include <hip/hip_bf16.h>
#include <math.h>

typedef _Float16 hf;
typedef __attribute__((ext_vector_type(16))) _Float16 v16h;
typedef __attribute__((ext_vector_type(8)))  _Float16 v8h;
typedef __attribute__((ext_vector_type(8)))  float    v8f;

#define HD __device__ __forceinline__

// ---------------- constants (match setup_inputs) ----------------
#define NB    256
#define NNW   32
#define NNODE 8192
#define N1    16384
#define N2    8192
#define E00   65536
#define E01   32768
#define E11   65536
#define E12   24576
#define FF    10
#define HDIM  128

HD float siluf(float x){ return x / (1.f + __expf(-x)); }
HD float sigmf(float x){ return 1.f / (1.f + __expf(-x)); }

HD v8f wmma16(v16h a, v16h b, v8f c){
  return __builtin_amdgcn_wmma_f32_16x16x32_f16(false, a, false, b, (short)0, c, false, false);
}

HD v16h cat8(v8h lo, v8h hi){
  return __builtin_shufflevector(lo, hi, 0,1,2,3,4,5,6,7,8,9,10,11,12,13,14,15);
}

// 8 consecutive f32 -> v8h (two global_load_b128 + packed cvt)
HD v8h cvt8(const float* __restrict__ p){
  float4 a = *(const float4*)p;
  float4 b = *(const float4*)(p + 4);
  v8h r;
  r[0]=(hf)a.x; r[1]=(hf)a.y; r[2]=(hf)a.z; r[3]=(hf)a.w;
  r[4]=(hf)b.x; r[5]=(hf)b.y; r[6]=(hf)b.z; r[7]=(hf)b.w;
  return r;
}

// B fragment: two runs of 8 f16 at p and p+16 from pre-transposed weights (two 16B loads)
HD v16h loadB(const hf* __restrict__ p){
  v8h lo = *(const v8h*)p;
  v8h hi = *(const v8h*)(p + 16);
  return cat8(lo, hi);
}

// A fragment run for message state [xs | xr | inv(pad)] (run of 8 at k0, k0 mult of 8)
HD v8h runA_msg(const float* __restrict__ xs_row, const float* __restrict__ xr_row,
                const float* __restrict__ inv_row, int inv_dim, int k0){
  if (k0 < 128) return cvt8(xs_row + k0);
  if (k0 < 256) return cvt8(xr_row + (k0 - 128));
  v8h r;
  #pragma unroll
  for (int t = 0; t < 8; ++t) {
    int kk = k0 - 256 + t;
    r[t] = (kk < inv_dim) ? (hf)inv_row[kk] : (hf)0.f;
  }
  return r;
}

// A fragment run for update state [x | mes...]
HD v8h runA_upd(const float* __restrict__ x_row, const float* __restrict__ m0r,
                const float* __restrict__ m1r, const float* __restrict__ m2r, int k0){
  const float* base;
  if (k0 < 128) base = x_row + k0;
  else {
    int j = (k0 >> 7) - 1;
    const float* mr = (j == 0) ? m0r : ((j == 1) ? m1r : m2r);
    base = mr + (k0 & 127);
  }
  return cvt8(base);
}

HD v8h runA_lds(const hf* __restrict__ row, int k0){
  return *(const v8h*)(row + k0);
}

HD float angf(const float* u, const float* v){
  float du = sqrtf(u[0]*u[0] + u[1]*u[1] + u[2]*u[2]);
  float dv = sqrtf(v[0]*v[0] + v[1]*v[1] + v[2]*v[2]);
  float dt = u[0]*v[0] + u[1]*v[1] + u[2]*v[2];
  float c  = dt / (du * dv + 1e-6f);
  c = fminf(fmaxf(c, -1.f + 1e-6f), 1.f - 1e-6f);
  return acosf(c);
}

// dot of v8h with 8 contiguous f32 weights
HD float dot8(v8h m, const float* __restrict__ w){
  float4 wa = *(const float4*)w;
  float4 wb = *(const float4*)(w + 4);
  return (float)m[0]*wa.x + (float)m[1]*wa.y + (float)m[2]*wa.z + (float)m[3]*wa.w
       + (float)m[4]*wb.x + (float)m[5]*wb.y + (float)m[6]*wb.z + (float)m[7]*wb.w;
}

// ============ weight pre-transpose/convert: w[K,N] f32 -> wt[128][Kp] f16 ======
__global__ void k_wconv(const float* __restrict__ w, hf* __restrict__ wt,
                        int K, int N, int Kp){
  int idx = blockIdx.x * blockDim.x + threadIdx.x;
  if (idx >= 128 * Kp) return;
  int n = idx / Kp, k = idx % Kp;
  wt[idx] = (k < K && n < N) ? (hf)w[(size_t)k * N + n] : (hf)0.f;
}

// ================= index precompute: x_ind = x + NN*batch =================
__global__ void k_ids(const int* __restrict__ base, const int* __restrict__ batch,
                      int* __restrict__ xind, int n, int arity, int nn){
  int i = blockIdx.x * blockDim.x + threadIdx.x;
  if (i >= n * arity) return;
  xind[i] = base[i] + nn * batch[i / arity];
}

// ================= feature embedding: cdist-of-frames -> 100 -> 128 =======
__global__ __launch_bounds__(128) void k_embed(
    const float* __restrict__ loc, const float* __restrict__ vel,
    const int* __restrict__ xind, int arity,
    const float* __restrict__ wemb, const float* __restrict__ bemb,
    float* __restrict__ x, float* __restrict__ pos_out, float* __restrict__ vel_out, int n)
{
  int lane = threadIdx.x & 31, wv = threadIdx.x >> 5;
  int node = blockIdx.x * 4 + wv;
  bool act = node < n;
  int nd = act ? node : 0;
  __shared__ float pos_s[4][32];
  __shared__ float pc_s[4][32];
  __shared__ float cd_s[4][100];
  if (lane < 30) {
    float ps = 0.f, vs = 0.f;
    for (int j = 0; j < arity; ++j) {
      int id = xind[(size_t)nd * arity + j];
      ps += loc[(size_t)id * 30 + lane];
      vs += vel[(size_t)id * 30 + lane];
    }
    ps /= (float)arity; vs /= (float)arity;
    pos_s[wv][lane] = ps;
    if (pos_out && act) { pos_out[(size_t)nd * 30 + lane] = ps; vel_out[(size_t)nd * 30 + lane] = vs; }
  }
  __syncthreads();
  if (lane < 30) {
    int d = lane % 3;
    float mn = 0.f;
    for (int f = 0; f < FF; ++f) mn += pos_s[wv][f * 3 + d];
    mn *= (1.f / FF);
    pc_s[wv][lane] = pos_s[wv][lane] - mn;
  }
  __syncthreads();
  for (int i = lane; i < 100; i += 32) {
    int f1 = i / 10, f2 = i % 10;
    float s = 0.f;
    for (int d = 0; d < 3; ++d) {
      float df = pc_s[wv][f1 * 3 + d] - pc_s[wv][f2 * 3 + d];
      s += df * df;
    }
    cd_s[wv][i] = sqrtf(s);
  }
  __syncthreads();
  if (act) {
    for (int o = lane; o < HDIM; o += 32) {
      float acc = bemb[o];
      for (int k = 0; k < 100; ++k) acc += cd_s[wv][k] * wemb[(size_t)k * HDIM + o];
      x[(size_t)nd * HDIM + o] = acc;
    }
  }
}

// ================= invariants ============================================
__global__ void k_inv00(const float* __restrict__ pos0, const int* __restrict__ as,
                        const int* __restrict__ ar, float* __restrict__ inv, int E){
  int idx = blockIdx.x * blockDim.x + threadIdx.x;
  if (idx >= E * FF) return;
  int e = idx / FF, f = idx % FF;
  const float* ps = pos0 + (size_t)as[e] * 30 + f * 3;
  const float* pr = pos0 + (size_t)ar[e] * 30 + f * 3;
  float dx = ps[0]-pr[0], dy = ps[1]-pr[1], dz = ps[2]-pr[2];
  float* o = inv + (size_t)e * 30 + f * 3;
  o[0] = sqrtf(dx*dx + dy*dy + dz*dz); o[1] = 0.f; o[2] = 0.f;
}

__global__ void k_inv01(const float* __restrict__ pos0, const int* __restrict__ as,
                        const int* __restrict__ ar, const int* __restrict__ xind1,
                        float* __restrict__ inv, int E){
  int idx = blockIdx.x * blockDim.x + threadIdx.x;
  if (idx >= E * FF) return;
  int e = idx / FF, f = idx % FF;
  int s = as[e], r = ar[e];
  const float* pa = pos0 + (size_t)xind1[r*2]   * 30 + f * 3;
  const float* pb = pos0 + (size_t)xind1[r*2+1] * 30 + f * 3;
  const float* ps = pos0 + (size_t)s * 30 + f * 3;
  float ex = pa[0]-pb[0], ey = pa[1]-pb[1], ez = pa[2]-pb[2];
  float cx = (pa[0]+pb[0])*0.5f, cy = (pa[1]+pb[1])*0.5f, cz = (pa[2]+pb[2])*0.5f;
  float dx = ps[0]-cx, dy = ps[1]-cy, dz = ps[2]-cz;
  float* o = inv + (size_t)e * 30 + f * 3;
  o[0] = sqrtf(dx*dx+dy*dy+dz*dz); o[1] = 0.f; o[2] = sqrtf(ex*ex+ey*ey+ez*ez);
}

__global__ void k_inv11(const float* __restrict__ pos0, const int* __restrict__ as,
                        const int* __restrict__ ar, const int* __restrict__ xind1,
                        float* __restrict__ inv, int E){
  int idx = blockIdx.x * blockDim.x + threadIdx.x;
  if (idx >= E * FF) return;
  int e = idx / FF, f = idx % FF;
  int s = as[e], r = ar[e];
  const float* sa = pos0 + (size_t)xind1[s*2]   * 30 + f * 3;
  const float* sb = pos0 + (size_t)xind1[s*2+1] * 30 + f * 3;
  const float* ra = pos0 + (size_t)xind1[r*2]   * 30 + f * 3;
  const float* rb = pos0 + (size_t)xind1[r*2+1] * 30 + f * 3;
  float vs[3] = { sa[0]-sb[0], sa[1]-sb[1], sa[2]-sb[2] };
  float vr[3] = { ra[0]-rb[0], ra[1]-rb[1], ra[2]-rb[2] };
  float dx = (sa[0]+sb[0]-ra[0]-rb[0])*0.5f;
  float dy = (sa[1]+sb[1]-ra[1]-rb[1])*0.5f;
  float dz = (sa[2]+sb[2]-ra[2]-rb[2])*0.5f;
  float sn = sqrtf(vs[0]*vs[0]+vs[1]*vs[1]+vs[2]*vs[2]);
  float rn = sqrtf(vr[0]*vr[0]+vr[1]*vr[1]+vr[2]*vr[2]);
  float* o = inv + (size_t)e * 60 + f * 6;
  o[0] = sqrtf(dx*dx+dy*dy+dz*dz); o[1] = sn; o[2] = rn;
  o[3] = angf(vs, vr); o[4] = sn; o[5] = rn;
}

__global__ void k_inv12(const float* __restrict__ pos0, const int* __restrict__ as,
                        const int* __restrict__ ar, const int* __restrict__ xind1,
                        const int* __restrict__ xind2,
                        const int* __restrict__ i1, const int* __restrict__ i2,
                        const int* __restrict__ ia, float* __restrict__ inv, int E){
  int idx = blockIdx.x * blockDim.x + threadIdx.x;
  if (idx >= E * FF) return;
  int e = idx / FF, f = idx % FF;
  int s = as[e], r = ar[e];
  const float* sa = pos0 + (size_t)xind1[s*2]   * 30 + f * 3;
  const float* sb = pos0 + (size_t)xind1[s*2+1] * 30 + f * 3;
  const float* ta = pos0 + (size_t)xind2[r*3]   * 30 + f * 3;
  const float* tb = pos0 + (size_t)xind2[r*3+1] * 30 + f * 3;
  const float* tc = pos0 + (size_t)xind2[r*3+2] * 30 + f * 3;
  float cs[3] = { (sa[0]+sb[0])*0.5f, (sa[1]+sb[1])*0.5f, (sa[2]+sb[2])*0.5f };
  float ct[3] = { (ta[0]+tb[0]+tc[0])/3.f, (ta[1]+tb[1]+tc[1])/3.f, (ta[2]+tb[2]+tc[2])/3.f };
  float u[3] = { ta[0]-tb[0], ta[1]-tb[1], ta[2]-tb[2] };
  float w[3] = { ta[0]-tc[0], ta[1]-tc[1], ta[2]-tc[2] };
  float cr[3] = { u[1]*w[2]-u[2]*w[1], u[2]*w[0]-u[0]*w[2], u[0]*w[1]-u[1]*w[0] };
  float dx = cs[0]-ct[0], dy = cs[1]-ct[1], dz = cs[2]-ct[2];
  float ex = sa[0]-sb[0], ey = sa[1]-sb[1], ez = sa[2]-sb[2];
  const float* p1 = pos0 + (size_t)i1[e] * 30 + f * 3;
  const float* p2 = pos0 + (size_t)i2[e] * 30 + f * 3;
  const float* pA = pos0 + (size_t)ia[e] * 30 + f * 3;
  float v1[3] = { p1[0]-pA[0], p1[1]-pA[1], p1[2]-pA[2] };
  float v2[3] = { p2[0]-pA[0], p2[1]-pA[1], p2[2]-pA[2] };
  float bv[3] = { p1[0]-p2[0], p1[1]-p2[1], p1[2]-p2[2] };
  float* o = inv + (size_t)e * 60 + f * 6;
  o[0] = sqrtf(dx*dx+dy*dy+dz*dz);
  o[1] = sqrtf(ex*ex+ey*ey+ez*ez);
  o[2] = 0.5f * sqrtf(cr[0]*cr[0]+cr[1]*cr[1]+cr[2]*cr[2]);
  o[3] = angf(v1, bv); o[4] = angf(v2, bv); o[5] = angf(v1, v2);
}

// ================= WMMA edge-message kernel (templated on inv width) ======
template<int INVD>
__global__ __launch_bounds__(128) void k_message(
    const float* __restrict__ xs, const float* __restrict__ xr,
    const float* __restrict__ inv,
    const int* __restrict__ src, const int* __restrict__ rcv,
    const hf* __restrict__ w1t, const float* __restrict__ b1,
    const hf* __restrict__ w2t, const float* __restrict__ b2,
    const float* __restrict__ wi, const float* __restrict__ bi,
    const float* __restrict__ wp, const float* __restrict__ bp,
    float* __restrict__ mes, float* __restrict__ pos_scalar, int E)
{
  constexpr int K  = 256 + INVD;
  constexpr int KT = (K + 31) >> 5;
  constexpr int KP = KT * 32;
  const int lane = threadIdx.x & 31;
  const int wv   = threadIdx.x >> 5;
  const int tile = blockIdx.x * 4 + wv;
  const bool act = (tile * 16) < E;
  const int e0 = act ? tile * 16 : 0;

  __shared__ hf    h_lds[4][16][HDIM];
  __shared__ hf    m_lds[4][16][HDIM];
  __shared__ float w_lds[4][16];
  __shared__ int   r_lds[4][16];

  const int row = lane & 15, col = lane & 15, half = lane >> 4;
  int erow = e0 + row; if (erow >= E) erow = E - 1;
  const int si = src[erow], ri = rcv[erow];
  if (lane < 16) r_lds[wv][row] = (act && (e0 + row) < E) ? ri : -1;
  const float* xs_row  = xs  + (size_t)si * HDIM;
  const float* xr_row  = xr  + (size_t)ri * HDIM;
  const float* inv_row = inv + (size_t)erow * INVD;

  // hoisted B row pointers (register-resident after unroll)
  const hf* wb1[8];
  const hf* wb2[8];
  #pragma unroll
  for (int nt = 0; nt < 8; ++nt) {
    wb1[nt] = w1t + (size_t)(nt * 16 + col) * KP + 8 * half;
    wb2[nt] = w2t + (size_t)(nt * 16 + col) * HDIM + 8 * half;
  }

  // GEMM1 + SiLU
  v8f acc[8];
  #pragma unroll
  for (int nt = 0; nt < 8; ++nt) {
    float bz = b1[nt * 16 + col];
    #pragma unroll
    for (int j = 0; j < 8; ++j) acc[nt][j] = bz;
  }
  #pragma unroll
  for (int kt = 0; kt < KT; ++kt) {
    int k0 = kt * 32 + 8 * half;
    v16h af = cat8(runA_msg(xs_row, xr_row, inv_row, INVD, k0),
                   runA_msg(xs_row, xr_row, inv_row, INVD, k0 + 16));
    #pragma unroll
    for (int nt = 0; nt < 8; ++nt) {
      v16h bf = loadB(wb1[nt] + kt * 32);
      acc[nt] = wmma16(af, bf, acc[nt]);
    }
  }
  #pragma unroll
  for (int nt = 0; nt < 8; ++nt)
    #pragma unroll
    for (int j = 0; j < 8; ++j)
      h_lds[wv][j + 8 * half][nt * 16 + col] = (hf)siluf(acc[nt][j]);
  __syncthreads();

  // GEMM2 + SiLU
  #pragma unroll
  for (int nt = 0; nt < 8; ++nt) {
    float bz = b2[nt * 16 + col];
    #pragma unroll
    for (int j = 0; j < 8; ++j) acc[nt][j] = bz;
  }
  #pragma unroll
  for (int kt = 0; kt < 4; ++kt) {
    int k0 = kt * 32 + 8 * half;
    v16h af = cat8(runA_lds(h_lds[wv][row], k0), runA_lds(h_lds[wv][row], k0 + 16));
    #pragma unroll
    for (int nt = 0; nt < 8; ++nt) {
      v16h bf = loadB(wb2[nt] + kt * 32);
      acc[nt] = wmma16(af, bf, acc[nt]);
    }
  }
  #pragma unroll
  for (int nt = 0; nt < 8; ++nt)
    #pragma unroll
    for (int j = 0; j < 8; ++j)
      m_lds[wv][j + 8 * half][nt * 16 + col] = (hf)siluf(acc[nt][j]);
  __syncthreads();

  // gate + optional pos scalar: all 32 lanes, half features each, shfl-combine
  {
    int er = lane & 15;
    float part = 0.f;
    #pragma unroll
    for (int t = 0; t < 8; ++t) {
      v8h mv = *(const v8h*)&m_lds[wv][er][half * 64 + t * 8];
      part += dot8(mv, wi + half * 64 + t * 8);
    }
    float tot = part + __shfl_xor(part, 16);
    if (lane < 16) w_lds[wv][er] = sigmf(tot + bi[0]);
    if (pos_scalar) {
      float pp = 0.f;
      #pragma unroll
      for (int t = 0; t < 8; ++t) {
        v8h mv = *(const v8h*)&m_lds[wv][er][half * 64 + t * 8];
        pp += dot8(mv, wp + half * 64 + t * 8);
      }
      float pt = pp + __shfl_xor(pp, 16) + bp[0];
      int e = e0 + er;
      if (act && e < E && lane < 16) pos_scalar[e] = pt;
    }
  }
  __syncthreads();

  // scatter: segment_sum via f32 atomics (receivers cached in LDS)
  for (int idx = lane; idx < 16 * HDIM; idx += 32) {
    int er = idx >> 7, f = idx & 127;
    int rr = r_lds[wv][er];
    if (rr >= 0)
      atomicAdd(&mes[(size_t)rr * HDIM + f], (float)m_lds[wv][er][f] * w_lds[wv][er]);
  }
}

// ================= generic WMMA MLP (templated on K) ======================
template<int KP1>
__global__ __launch_bounds__(128) void k_mlp(
    const float* __restrict__ x,
    const float* __restrict__ m0, const float* __restrict__ m1p, const float* __restrict__ m2p,
    const hf* __restrict__ w1t, const float* __restrict__ b1,
    const hf* __restrict__ w2t, const float* __restrict__ b2,
    float* __restrict__ out, int n, int dout, int residual)
{
  constexpr int KT = KP1 >> 5;
  const int lane = threadIdx.x & 31;
  const int wv   = threadIdx.x >> 5;
  const int tile = blockIdx.x * 4 + wv;
  const bool act = (tile * 16) < n;
  const int n0 = act ? tile * 16 : 0;
  __shared__ hf h_lds[4][16][HDIM];
  const int row = lane & 15, col = lane & 15, half = lane >> 4;
  int nrow = n0 + row; if (nrow >= n) nrow = n - 1;
  const float* x_row = x + (size_t)nrow * HDIM;
  const float* m0r = m0 ? m0 + (size_t)nrow * HDIM : nullptr;
  const float* m1r = m1p ? m1p + (size_t)nrow * HDIM : nullptr;
  const float* m2r = m2p ? m2p + (size_t)nrow * HDIM : nullptr;

  const hf* wb1[8];
  const hf* wb2[8];
  #pragma unroll
  for (int nt = 0; nt < 8; ++nt) {
    wb1[nt] = w1t + (size_t)(nt * 16 + col) * KP1 + 8 * half;
    wb2[nt] = w2t + (size_t)(nt * 16 + col) * HDIM + 8 * half;
  }

  v8f acc[8];
  #pragma unroll
  for (int nt = 0; nt < 8; ++nt) {
    float bz = b1[nt * 16 + col];
    #pragma unroll
    for (int j = 0; j < 8; ++j) acc[nt][j] = bz;
  }
  #pragma unroll
  for (int kt = 0; kt < KT; ++kt) {
    int k0 = kt * 32 + 8 * half;
    v16h af = cat8(runA_upd(x_row, m0r, m1r, m2r, k0),
                   runA_upd(x_row, m0r, m1r, m2r, k0 + 16));
    #pragma unroll
    for (int nt = 0; nt < 8; ++nt) {
      v16h bf = loadB(wb1[nt] + kt * 32);
      acc[nt] = wmma16(af, bf, acc[nt]);
    }
  }
  #pragma unroll
  for (int nt = 0; nt < 8; ++nt)
    #pragma unroll
    for (int j = 0; j < 8; ++j)
      h_lds[wv][j + 8 * half][nt * 16 + col] = (hf)siluf(acc[nt][j]);
  __syncthreads();

  v8f acc2[8];
  #pragma unroll
  for (int nt = 0; nt < 8; ++nt) {
    int cg = nt * 16 + col;
    float bz = (cg < dout) ? b2[cg] : 0.f;
    #pragma unroll
    for (int j = 0; j < 8; ++j) acc2[nt][j] = bz;
  }
  #pragma unroll
  for (int kt = 0; kt < 4; ++kt) {
    int k0 = kt * 32 + 8 * half;
    v16h af = cat8(runA_lds(h_lds[wv][row], k0), runA_lds(h_lds[wv][row], k0 + 16));
    #pragma unroll
    for (int nt = 0; nt < 8; ++nt) {
      v16h bf = loadB(wb2[nt] + kt * 32);
      acc2[nt] = wmma16(af, bf, acc2[nt]);
    }
  }
  #pragma unroll
  for (int nt = 0; nt < 8; ++nt) {
    int cg = nt * 16 + col;
    if (cg >= dout) continue;
    #pragma unroll
    for (int j = 0; j < 8; ++j) {
      int nr = n0 + j + 8 * half;
      if (act && nr < n) {
        float v = acc2[nt][j];
        if (residual) v += x[(size_t)nr * HDIM + cg];
        out[(size_t)nr * dout + cg] = v;
      }
    }
  }
}

// ================= EGNN position / velocity updates =======================
__global__ void k_vcoef(const float* __restrict__ xn0, const float* __restrict__ wv,
                        const float* __restrict__ bv, float* __restrict__ vc, int n){
  int i = blockIdx.x * blockDim.x + threadIdx.x;
  if (i >= n) return;
  float s = bv[0];
  for (int k = 0; k < HDIM; k += 4) {
    float4 xv = *(const float4*)(xn0 + (size_t)i * HDIM + k);
    float4 wvv = *(const float4*)(wv + k);
    s += xv.x*wvv.x + xv.y*wvv.y + xv.z*wvv.z + xv.w*wvv.w;
  }
  vc[i] = s;
}

__global__ void k_posagg(const float* __restrict__ pos0, const float* __restrict__ pscal,
                         const int* __restrict__ as, const int* __restrict__ ar,
                         float* __restrict__ agg, float* __restrict__ cnt, int E){
  int idx = blockIdx.x * blockDim.x + threadIdx.x;
  if (idx >= E * 30) return;
  int e = idx / 30, c = idx % 30;
  int s = as[e], r = ar[e];
  float tr = (pos0[(size_t)s * 30 + c] - pos0[(size_t)r * 30 + c]) * pscal[e];
  atomicAdd(&agg[(size_t)r * 30 + c], tr);
  if (c == 0) atomicAdd(&cnt[r], 1.f);
}

__global__ void k_velpos(const float* __restrict__ vc, const float* __restrict__ agg,
                         const float* __restrict__ cnt, float* __restrict__ vel0,
                         float* __restrict__ pos0, int n){
  int idx = blockIdx.x * blockDim.x + threadIdx.x;
  if (idx >= n * 30) return;
  int i = idx / 30;
  float a = agg[idx] / fmaxf(cnt[i], 1.f);
  float v = vc[i] * vel0[idx] + a;
  vel0[idx] = v;
  pos0[idx] += v;
}

// ================= output head ===========================================
__global__ void k_out(const float* __restrict__ out30, const float* __restrict__ pos0,
                      const float* __restrict__ loc, const float* __restrict__ y,
                      float* __restrict__ dist, int n){
  int idx = blockIdx.x * blockDim.x + threadIdx.x;
  if (idx >= n * FF) return;
  int nd = idx / FF, f = idx % FF;
  float s = 0.f;
  for (int d = 0; d < 3; ++d) {
    size_t o = (size_t)nd * 30 + f * 3 + d;
    float lp = out30[o] * pos0[o] + loc[o];
    float df = lp - y[o];
    s += df * df;
  }
  dist[(size_t)nd * FF + f] = sqrtf(s);
}

__global__ void k_reduce(const float* __restrict__ dist, float* __restrict__ out){
  int b = blockIdx.x, t = threadIdx.x;
  __shared__ float s1[64], s2[64];
  float a = 0.f, fde = 0.f;
  for (int i = t; i < NNW * FF; i += 64) {
    int nl = i / FF, fr = i % FF;
    a += dist[(size_t)(b * NNW + nl) * FF + fr];
  }
  for (int i = t; i < NNW; i += 64)
    fde += dist[(size_t)(b * NNW + i) * FF + (FF - 1)];
  s1[t] = a; s2[t] = fde;
  __syncthreads();
  for (int st = 32; st > 0; st >>= 1) {
    if (t < st) { s1[t] += s1[t + st]; s2[t] += s2[t + st]; }
    __syncthreads();
  }
  if (t == 0) {
    out[1 + b]       = s1[0] / (float)(NNW * FF);
    out[1 + NB + b]  = s2[0] / (float)NNW;
  }
}

__global__ void k_ademean(float* __restrict__ out){
  __shared__ float s[NB];
  int t = threadIdx.x;
  s[t] = out[1 + t];
  __syncthreads();
  for (int st = NB / 2; st > 0; st >>= 1) {
    if (t < st) s[t] += s[t + st];
    __syncthreads();
  }
  if (t == 0) out[0] = s[0] / (float)NB;
}

// ========================= host side =====================================
static inline int cdiv_h(int a, int b){ return (a + b - 1) / b; }

extern "C" void kernel_launch(void* const* d_in, const int* in_sizes, int n_in,
                              void* d_out, int out_size, void* d_ws, size_t ws_size,
                              hipStream_t stream) {
  (void)in_sizes; (void)n_in; (void)out_size; (void)ws_size;
  const float* loc = (const float*)d_in[0];
  const float* vel = (const float*)d_in[1];
  const float* yin = (const float*)d_in[2];
  const int P = 3;  // params leaves: P .. P+381 (JAX pytree, sorted dict keys)
  auto Fp = [&](int i)->const float*{ return (const float*)d_in[i]; };
  const float* emb_b = Fp(P + 0);
  const float* emb_w = Fp(P + 1);
  const float* post_l1b = Fp(P + 366); const float* post_l1w = Fp(P + 367);
  const float* post_l2b = Fp(P + 368); const float* post_l2w = Fp(P + 369);
  const float* pre0_l1b = Fp(P + 370); const float* pre0_l1w = Fp(P + 371);
  const float* pre0_l2b = Fp(P + 372); const float* pre0_l2w = Fp(P + 373);
  const int* x_0    = (const int*)d_in[P + 382 + 0];
  const int* x_1    = (const int*)d_in[P + 382 + 1];
  const int* x_2    = (const int*)d_in[P + 382 + 2];
  const int* b_0    = (const int*)d_in[P + 382 + 3];
  const int* b_1    = (const int*)d_in[P + 382 + 4];
  const int* b_2    = (const int*)d_in[P + 382 + 5];
  const int* adj00  = (const int*)d_in[P + 382 + 7];
  const int* adj01  = (const int*)d_in[P + 382 + 8];
  const int* adj11  = (const int*)d_in[P + 382 + 9];
  const int* adj12  = (const int*)d_in[P + 382 + 10];
  const int* inv12i = (const int*)d_in[P + 382 + 11];

  // ---- workspace carve ----
  size_t off = 0;
  auto carve = [&](size_t bytes)->char*{
    char* r = (char*)d_ws + off;
    off += (bytes + 255) & ~(size_t)255;
    return r;
  };
  int*   xind0 = (int*)carve((size_t)NNODE * 4);
  int*   xind1 = (int*)carve((size_t)N1 * 2 * 4);
  int*   xind2 = (int*)carve((size_t)N2 * 3 * 4);
  float* pos0  = (float*)carve((size_t)NNODE * 30 * 4);
  float* vel0  = (float*)carve((size_t)NNODE * 30 * 4);
  float* x0    = (float*)carve((size_t)NNODE * HDIM * 4);
  float* x0b   = (float*)carve((size_t)NNODE * HDIM * 4);
  float* x1    = (float*)carve((size_t)N1 * HDIM * 4);
  float* x1b   = (float*)carve((size_t)N1 * HDIM * 4);
  float* x2    = (float*)carve((size_t)N2 * HDIM * 4);
  float* x2b   = (float*)carve((size_t)N2 * HDIM * 4);
  float* mes00 = (float*)carve((size_t)NNODE * HDIM * 4);
  float* mes10 = (float*)carve((size_t)NNODE * HDIM * 4);
  float* mes01 = (float*)carve((size_t)N1 * HDIM * 4);
  float* mes11 = (float*)carve((size_t)N1 * HDIM * 4);
  float* mes21 = (float*)carve((size_t)N1 * HDIM * 4);
  float* mes12 = (float*)carve((size_t)N2 * HDIM * 4);
  float* inv00 = (float*)carve((size_t)E00 * 30 * 4);
  float* inv01 = (float*)carve((size_t)E01 * 30 * 4);
  float* inv11 = (float*)carve((size_t)E11 * 60 * 4);
  float* inv12 = (float*)carve((size_t)E12 * 60 * 4);
  float* pscal = (float*)carve((size_t)E00 * 4);
  float* agg   = (float*)carve((size_t)NNODE * 30 * 4);
  float* cnt   = (float*)carve((size_t)NNODE * 4);
  float* vc    = (float*)carve((size_t)NNODE * 4);
  float* xp0   = (float*)carve((size_t)NNODE * HDIM * 4);
  float* out30 = (float*)carve((size_t)NNODE * 30 * 4);
  float* dist  = (float*)carve((size_t)NNODE * FF * 4);

  // ---- f16 transposed weight caches ----
  static const int m1Kp[6] = {288, 288, 288, 320, 320, 320};
  static const int m1K [6] = {286, 286, 286, 316, 316, 316};
  static const int u1K [3] = {384, 512, 256};
  hf* m1t[7][6]; hf* m2t[7][6]; hf* u1t[7][3]; hf* u2t[7][3];
  for (int l = 0; l < 7; ++l) {
    for (int a = 0; a < 6; ++a) {
      m1t[l][a] = (hf*)carve((size_t)HDIM * m1Kp[a] * 2);
      m2t[l][a] = (hf*)carve((size_t)HDIM * HDIM * 2);
    }
    for (int d = 0; d < 3; ++d) {
      u1t[l][d] = (hf*)carve((size_t)HDIM * u1K[d] * 2);
      u2t[l][d] = (hf*)carve((size_t)HDIM * HDIM * 2);
    }
  }
  hf* pre1t  = (hf*)carve((size_t)HDIM * HDIM * 2);
  hf* pre2t  = (hf*)carve((size_t)HDIM * HDIM * 2);
  hf* post1t = (hf*)carve((size_t)HDIM * HDIM * 2);
  hf* post2t = (hf*)carve((size_t)HDIM * HDIM * 2);

  auto conv = [&](const float* w, hf* wt, int K, int N, int Kp){
    int total = 128 * Kp;
    k_wconv<<<cdiv_h(total, 256), 256, 0, stream>>>(w, wt, K, N, Kp);
  };
  for (int l = 0; l < 7; ++l) {
    const int L = P + 2 + l * 52;
    for (int a = 0; a < 6; ++a) {
      conv(Fp(L + a * 6 + 3), m1t[l][a], m1K[a], HDIM, m1Kp[a]);  // m1.w
      conv(Fp(L + a * 6 + 5), m2t[l][a], HDIM, HDIM, HDIM);        // m2.w
    }
    for (int d = 0; d < 3; ++d) {
      conv(Fp(L + 38 + d * 4 + 1), u1t[l][d], u1K[d], HDIM, u1K[d]); // upd l1.w
      conv(Fp(L + 38 + d * 4 + 3), u2t[l][d], HDIM, HDIM, HDIM);     // upd l2.w
    }
  }
  conv(pre0_l1w, pre1t, HDIM, HDIM, HDIM);
  conv(pre0_l2w, pre2t, HDIM, HDIM, HDIM);
  conv(post_l1w, post1t, HDIM, HDIM, HDIM);
  conv(post_l2w, post2t, HDIM, 30, HDIM);

  // ---- indices + initial features ----
  k_ids<<<cdiv_h(NNODE, 256), 256, 0, stream>>>(x_0, b_0, xind0, NNODE, 1, NNW);
  k_ids<<<cdiv_h(N1 * 2, 256), 256, 0, stream>>>(x_1, b_1, xind1, N1, 2, NNW);
  k_ids<<<cdiv_h(N2 * 3, 256), 256, 0, stream>>>(x_2, b_2, xind2, N2, 3, NNW);
  k_embed<<<cdiv_h(NNODE, 4), 128, 0, stream>>>(loc, vel, xind0, 1, emb_w, emb_b, x0, pos0, vel0, NNODE);
  k_embed<<<cdiv_h(N1, 4), 128, 0, stream>>>(loc, vel, xind1, 2, emb_w, emb_b, x1, nullptr, nullptr, N1);
  k_embed<<<cdiv_h(N2, 4), 128, 0, stream>>>(loc, vel, xind2, 3, emb_w, emb_b, x2, nullptr, nullptr, N2);

  float *cx0 = x0, *cx1 = x1, *cx2 = x2, *nx0 = x0b, *nx1 = x1b, *nx2 = x2b;

  for (int l = 0; l < 7; ++l) {
    const int L = P + 2 + l * 52;
    auto MPb = [&](int ai, int which)->const float*{ return Fp(L + ai * 6 + which); };
    const float* pos_b = Fp(L + 36); const float* pos_w = Fp(L + 37);
    const float* vel_b = Fp(L + 50); const float* vel_w = Fp(L + 51);

    k_inv00<<<cdiv_h(E00 * FF, 256), 256, 0, stream>>>(pos0, adj00, adj00 + E00, inv00, E00);
    k_inv01<<<cdiv_h(E01 * FF, 256), 256, 0, stream>>>(pos0, adj01, adj01 + E01, xind1, inv01, E01);
    k_inv11<<<cdiv_h(E11 * FF, 256), 256, 0, stream>>>(pos0, adj11, adj11 + E11, xind1, inv11, E11);
    k_inv12<<<cdiv_h(E12 * FF, 256), 256, 0, stream>>>(pos0, adj12, adj12 + E12, xind1, xind2,
                                                       inv12i, inv12i + E12, inv12i + 2 * E12, inv12, E12);
    hipMemsetAsync(mes00, 0, (size_t)NNODE * HDIM * 4, stream);
    hipMemsetAsync(mes10, 0, (size_t)NNODE * HDIM * 4, stream);
    hipMemsetAsync(mes01, 0, (size_t)N1 * HDIM * 4, stream);
    hipMemsetAsync(mes11, 0, (size_t)N1 * HDIM * 4, stream);
    hipMemsetAsync(mes21, 0, (size_t)N1 * HDIM * 4, stream);
    hipMemsetAsync(mes12, 0, (size_t)N2 * HDIM * 4, stream);
    hipMemsetAsync(agg,   0, (size_t)NNODE * 30 * 4, stream);
    hipMemsetAsync(cnt,   0, (size_t)NNODE * 4, stream);

    struct MsgCfg { const float *xs, *xr, *inv; const int *src, *rcv; float* mes;
                    float* ps; int E, invd, ai; };
    MsgCfg cfgs[6] = {
      { cx0, cx0, inv00, adj00,       adj00 + E00, mes00, pscal,   E00, 30, 0 },
      { cx0, cx1, inv01, adj01,       adj01 + E01, mes01, nullptr, E01, 30, 1 },
      { cx1, cx0, inv01, adj01 + E01, adj01,       mes10, nullptr, E01, 30, 2 },
      { cx1, cx1, inv11, adj11,       adj11 + E11, mes11, nullptr, E11, 60, 3 },
      { cx1, cx2, inv12, adj12,       adj12 + E12, mes12, nullptr, E12, 60, 4 },
      { cx2, cx1, inv12, adj12 + E12, adj12,       mes21, nullptr, E12, 60, 5 },
    };
    for (int c = 0; c < 6; ++c) {
      const MsgCfg& m = cfgs[c];
      int blocks = cdiv_h(cdiv_h(m.E, 16), 4);
      if (m.invd == 30) {
        k_message<30><<<blocks, 128, 0, stream>>>(
            m.xs, m.xr, m.inv, m.src, m.rcv,
            m1t[l][m.ai], MPb(m.ai, 2), m2t[l][m.ai], MPb(m.ai, 4),
            MPb(m.ai, 1), MPb(m.ai, 0), pos_w, pos_b, m.mes, m.ps, m.E);
      } else {
        k_message<60><<<blocks, 128, 0, stream>>>(
            m.xs, m.xr, m.inv, m.src, m.rcv,
            m1t[l][m.ai], MPb(m.ai, 2), m2t[l][m.ai], MPb(m.ai, 4),
            MPb(m.ai, 1), MPb(m.ai, 0), pos_w, pos_b, m.mes, m.ps, m.E);
      }
    }

    auto Ub = [&](int d, int which)->const float*{ return Fp(L + 38 + d * 4 + which); };
    k_mlp<384><<<cdiv_h(cdiv_h(NNODE, 16), 4), 128, 0, stream>>>(
        cx0, mes00, mes10, nullptr, u1t[l][0], Ub(0, 0), u2t[l][0], Ub(0, 2),
        nx0, NNODE, HDIM, 1);
    k_mlp<512><<<cdiv_h(cdiv_h(N1, 16), 4), 128, 0, stream>>>(
        cx1, mes01, mes11, mes21, u1t[l][1], Ub(1, 0), u2t[l][1], Ub(1, 2),
        nx1, N1, HDIM, 1);
    k_mlp<256><<<cdiv_h(cdiv_h(N2, 16), 4), 128, 0, stream>>>(
        cx2, mes12, nullptr, nullptr, u1t[l][2], Ub(2, 0), u2t[l][2], Ub(2, 2),
        nx2, N2, HDIM, 1);

    k_vcoef<<<cdiv_h(NNODE, 256), 256, 0, stream>>>(nx0, vel_w, vel_b, vc, NNODE);
    k_posagg<<<cdiv_h(E00 * 30, 256), 256, 0, stream>>>(pos0, pscal, adj00, adj00 + E00, agg, cnt, E00);
    k_velpos<<<cdiv_h(NNODE * 30, 256), 256, 0, stream>>>(vc, agg, cnt, vel0, pos0, NNODE);

    { float* t = cx0; cx0 = nx0; nx0 = t; }
    { float* t = cx1; cx1 = nx1; nx1 = t; }
    { float* t = cx2; cx2 = nx2; nx2 = t; }
  }

  k_mlp<128><<<cdiv_h(cdiv_h(NNODE, 16), 4), 128, 0, stream>>>(
      cx0, nullptr, nullptr, nullptr, pre1t, pre0_l1b, pre2t, pre0_l2b,
      xp0, NNODE, HDIM, 0);
  k_mlp<128><<<cdiv_h(cdiv_h(NNODE, 16), 4), 128, 0, stream>>>(
      xp0, nullptr, nullptr, nullptr, post1t, post_l1b, post2t, post_l2b,
      out30, NNODE, 30, 0);
  k_out<<<cdiv_h(NNODE * FF, 256), 256, 0, stream>>>(out30, pos0, loc, yin, dist, NNODE);
  k_reduce<<<NB, 64, 0, stream>>>(dist, (float*)d_out);
  k_ademean<<<1, NB, 0, stream>>>((float*)d_out);
}
